// torch_GRL_61615600828815
// MI455X (gfx1250) — matrix operand order repaired and checked by reading, not compiled, not run on hardware
//
#include <hip/hip_runtime.h>

// ---------------------------------------------------------------------------
// GNN pipeline for MI455X (gfx1250, wave32).
//
// Roofline: the 1 GiB dense 0/1 adjacency must be fully read once (for deg
// AND to find nonzeros). The deg pass repacks it as a 32 MB bitmask
// (symmetric matrix -> column bits stored as row bits, no transpose), so the
// WMMA SpMM pass streams 32 MB instead of re-reading 1 GiB: total HBM
// ~1.2 GiB (~50us @ 23.3 TB/s).
//
// The big matmul  S = A_hat^T @ (dinv * (h @ wg))  (16384x16384x128) runs on
// v_wmma_f32_16x16x32_bf16: A expands bits -> bf16 0/1 exactly (v_cndmask_b16
// pairs); the dense operand z is split into bf16 hi+lo and both products
// accumulate into one f32 accumulator (error ~2^-17, ~fp32 accurate).
// The j-sweep is split 8-ways across blocks (8192 waves -> ~8/SIMD) so that
// resident-wave interleaving hides the per-WMMA s_wait_loadcnt stalls the
// compiler's single-buffer B schedule leaves behind; partial f32 tiles are
// summed in a coalesced reduce kernel that applies the exact +I diagonal,
// dinv scaling, bias and relu.
// ---------------------------------------------------------------------------

#define NN 16384      // nodes
#define NW (NN / 32)  // bitmask words per row
#define JSPLIT 8      // j-dimension split of the SpMM

typedef __attribute__((ext_vector_type(16))) __bf16 v16bf;
typedef __attribute__((ext_vector_type(8)))  float  v8f;

union BF16x16 { uint4 u[2]; v16bf v; };

__device__ __forceinline__ v16bf load_bf16x16(const __bf16* p) {
  BF16x16 r;
  r.u[0] = *(const uint4*)(p);
  r.u[1] = *(const uint4*)(p + 8);
  return r.v;
}

// ---------------------------------------------------------------------------
// Generic small fp32 GEMM: Y = act(X[N,IN] @ W[IN,OUT] + B)
// block = OUT*RG threads, RG*RPT rows per block. W loads coalesced over
// cols, X loads uniform (broadcast) across the col-threads of a row group.
// ---------------------------------------------------------------------------
template <int IN, int OUT, int RG, int RPT, bool RELU>
__global__ __launch_bounds__(OUT * RG) void lin_kernel(
    const float* __restrict__ X, const float* __restrict__ W,
    const float* __restrict__ Bv, float* __restrict__ Y) {
  const int col   = threadIdx.x % OUT;
  const int rg    = threadIdx.x / OUT;
  const int rbase = blockIdx.x * (RG * RPT) + rg;

  float acc[RPT];
#pragma unroll
  for (int s = 0; s < RPT; ++s) acc[s] = Bv[col];

  for (int j = 0; j < IN; ++j) {
    const float w = W[j * OUT + col];
#pragma unroll
    for (int s = 0; s < RPT; ++s)
      acc[s] += X[(size_t)(rbase + s * RG) * IN + j] * w;
  }
#pragma unroll
  for (int s = 0; s < RPT; ++s) {
    float v = acc[s];
    if (RELU) v = fmaxf(v, 0.0f);
    Y[(size_t)(rbase + s * RG) * OUT + col] = v;
  }
}

// p1 = relu(concat(xd, h) @ wp1 + bp1), wp1 is [256,128]
__global__ __launch_bounds__(256) void concat_lin_kernel(
    const float* __restrict__ X1, const float* __restrict__ X2,
    const float* __restrict__ W, const float* __restrict__ Bv,
    float* __restrict__ Y) {
  const int col   = threadIdx.x & 127;
  const int rg    = threadIdx.x >> 7;          // 0..1
  const int rbase = blockIdx.x * 16 + rg;

  float acc[8];
#pragma unroll
  for (int s = 0; s < 8; ++s) acc[s] = Bv[col];

  for (int j = 0; j < 128; ++j) {
    const float w = W[j * 128 + col];
#pragma unroll
    for (int s = 0; s < 8; ++s)
      acc[s] += X1[(size_t)(rbase + 2 * s) * 128 + j] * w;
  }
  for (int j = 0; j < 128; ++j) {
    const float w = W[(128 + j) * 128 + col];
#pragma unroll
    for (int s = 0; s < 8; ++s)
      acc[s] += X2[(size_t)(rbase + 2 * s) * 128 + j] * w;
  }
#pragma unroll
  for (int s = 0; s < 8; ++s)
    Y[(size_t)(rbase + 2 * s) * 128 + col] = fmaxf(acc[s], 0.0f);
}

// ---------------------------------------------------------------------------
// deg pass (the only full 1 GiB read): deterministic two-pass column sum of
// adj, fused with bit-packing. Thread owns column c; since adj is symmetric,
// column-c bits == row-c bits, so it writes the row-major bitmask directly
// (16 consecutive words = four uint4 stores per 512-row chunk).
// ---------------------------------------------------------------------------
__global__ __launch_bounds__(256) void colsum_pack_kernel(
    const float* __restrict__ adj, float* __restrict__ partial,
    unsigned* __restrict__ bits) {
  const int c  = blockIdx.x * 256 + threadIdx.x;
  const int r0 = blockIdx.y * 512;
  float s = 0.0f;

  for (int q = 0; q < 4; ++q) {        // 4 quads of 128 rows
    unsigned w[4];
#pragma unroll
    for (int g = 0; g < 4; ++g) {
      unsigned ww = 0u;
#pragma unroll
      for (int b = 0; b < 32; ++b) {
        const float v = adj[(size_t)(r0 + q * 128 + g * 32 + b) * NN + c];
        s += v;
        ww |= (v != 0.0f ? 1u : 0u) << b;
      }
      w[g] = ww;
    }
    *(uint4*)(bits + (size_t)c * NW + (r0 >> 5) + q * 4) =
        make_uint4(w[0], w[1], w[2], w[3]);
  }
  partial[(size_t)blockIdx.y * NN + c] = s;
}

__global__ __launch_bounds__(256) void dinv_kernel(
    const float* __restrict__ partial, float* __restrict__ dinv) {
  const int c = blockIdx.x * 256 + threadIdx.x;
  float s = 1.0f;  // +I self loop
#pragma unroll
  for (int k = 0; k < 32; ++k) s += partial[(size_t)k * NN + c];
  dinv[c] = rsqrtf(s);  // deg >= 1 always
}

// ---------------------------------------------------------------------------
// z = dinv * (h @ wg): keep f32 copy (row major, for diagonal add) and
// bf16 hi/lo split stored TRANSPOSED [128, NN] so the SpMM B-fragment
// (16 K-consecutive elements per lane) is 32 contiguous bytes.
// ---------------------------------------------------------------------------
__global__ __launch_bounds__(256) void zgen_kernel(
    const float* __restrict__ h, const float* __restrict__ wg,
    const float* __restrict__ dinv, float* __restrict__ zf,
    __bf16* __restrict__ zhiT, __bf16* __restrict__ zloT) {
  const int col   = threadIdx.x & 127;
  const int rg    = threadIdx.x >> 7;
  const int rbase = blockIdx.x * 16 + rg;

  float acc[8] = {0.f, 0.f, 0.f, 0.f, 0.f, 0.f, 0.f, 0.f};
  for (int j = 0; j < 128; ++j) {
    const float w = wg[j * 128 + col];
#pragma unroll
    for (int s = 0; s < 8; ++s)
      acc[s] += h[(size_t)(rbase + 2 * s) * 128 + j] * w;
  }
#pragma unroll
  for (int s = 0; s < 8; ++s) {
    const int row  = rbase + 2 * s;
    const float zv = dinv[row] * acc[s];
    zf[(size_t)row * 128 + col] = zv;
    const __bf16 hi = (__bf16)zv;
    const __bf16 lo = (__bf16)(zv - (float)hi);
    zhiT[(size_t)col * NN + row] = hi;
    zloT[(size_t)col * NN + row] = lo;
  }
}

// ---------------------------------------------------------------------------
// SpMM via WMMA bf16, partial over a j-chunk:
//   part[jsplit] = adj[rows, jchunk] @ z[jchunk, :]
// blockDim = 64 (2 waves); each wave owns a 16-row x 128-col f32 tile and
// sweeps NN/JSPLIT columns in steps of K=32. A fragment comes from ONE u32
// bitmask word per lane (expanded to bf16 0/1 with v_cndmask_b16).
// ---------------------------------------------------------------------------
__global__ __launch_bounds__(64) void spmm_wmma_kernel(
    const unsigned* __restrict__ bits, const __bf16* __restrict__ zhiT,
    const __bf16* __restrict__ zloT, float* __restrict__ part) {
  const int wave = threadIdx.x >> 5;
  const int lane = threadIdx.x & 31;
  const int half = lane >> 4;   // 0: lanes 0-15, 1: lanes 16-31
  const int mrow = lane & 15;
  const int r0   = blockIdx.x * 32 + wave * 16;
  const int j0   = blockIdx.y * (NN / JSPLIT);

  // A fragment (16x32 bf16, MxK): lane = M row (r0+mrow);
  // elems 0..15 <-> K = half*16 + 0..15
  const unsigned* arowbits = bits + (size_t)(r0 + mrow) * NW;
  const int kA = half * 16;
  // B fragment (32x16 bf16, KxN): lane -> N = mrow, K = half*16 + e (e=0..15)
  const int kB = half << 4;

  const __bf16 onebf  = (__bf16)1.0f;
  const __bf16 zerobf = (__bf16)0.0f;

  v8f acc[8];
#pragma unroll
  for (int t = 0; t < 8; ++t)
#pragma unroll
    for (int v = 0; v < 8; ++v) acc[t][v] = 0.0f;

  for (int jb = j0; jb < j0 + NN / JSPLIT; jb += 32) {
    const unsigned w = arowbits[jb >> 5];
    v16bf a;
#pragma unroll
    for (int e = 0; e < 8; ++e) {
      a[e]     = ((w >> (kA + e)) & 1u)      ? onebf : zerobf;
      a[8 + e] = ((w >> (kA + 8 + e)) & 1u) ? onebf : zerobf;
    }

    v16bf bfrag[8];
#pragma unroll
    for (int t = 0; t < 8; ++t)
      bfrag[t] = load_bf16x16(zhiT + (size_t)(t * 16 + mrow) * NN + jb + kB);
#pragma unroll
    for (int t = 0; t < 8; ++t)
      acc[t] = __builtin_amdgcn_wmma_f32_16x16x32_bf16(
          false, a, false, bfrag[t], (short)0, acc[t], false, false);
#pragma unroll
    for (int t = 0; t < 8; ++t)
      bfrag[t] = load_bf16x16(zloT + (size_t)(t * 16 + mrow) * NN + jb + kB);
#pragma unroll
    for (int t = 0; t < 8; ++t)
      acc[t] = __builtin_amdgcn_wmma_f32_16x16x32_bf16(
          false, a, false, bfrag[t], (short)0, acc[t], false, false);
  }

  // Store raw partial tile. C/D layout -> lane half picks M=v or v+8, N=mrow.
  float* pbase = part + (size_t)blockIdx.y * NN * 128;
#pragma unroll
  for (int t = 0; t < 8; ++t) {
    const int col = t * 16 + mrow;
#pragma unroll
    for (int v = 0; v < 8; ++v) {
      const int row = r0 + v + 8 * half;
      pbase[(size_t)row * 128 + col] = acc[t][v];
    }
  }
}

// xg = relu(dinv[row] * (sum_j part[j] + z[row]) + bg[col]), coalesced.
__global__ __launch_bounds__(256) void spmm_reduce_kernel(
    const float* __restrict__ part, const float* __restrict__ zf,
    const float* __restrict__ dinv, const float* __restrict__ bg,
    float* __restrict__ xg) {
  const size_t i = (size_t)blockIdx.x * 256 + threadIdx.x;
  const int row = (int)(i >> 7);
  const int col = (int)(i & 127);
  float s = zf[i];
#pragma unroll
  for (int k = 0; k < JSPLIT; ++k) s += part[(size_t)k * NN * 128 + i];
  xg[i] = fmaxf(dinv[row] * s + bg[col], 0.0f);
}

// out = p2 @ wo + bo  ([N,64] x [64,5])
__global__ __launch_bounds__(256) void out_kernel(
    const float* __restrict__ p2, const float* __restrict__ wo,
    const float* __restrict__ bo, float* __restrict__ out) {
  const int i = blockIdx.x * 256 + threadIdx.x;
  float acc[5];
#pragma unroll
  for (int a = 0; a < 5; ++a) acc[a] = bo[a];
  for (int j = 0; j < 64; ++j) {
    const float p = p2[(size_t)i * 64 + j];
#pragma unroll
    for (int a = 0; a < 5; ++a) acc[a] += p * wo[j * 5 + a];
  }
#pragma unroll
  for (int a = 0; a < 5; ++a) out[(size_t)i * 5 + a] = acc[a];
}

// ---------------------------------------------------------------------------
extern "C" void kernel_launch(void* const* d_in, const int* in_sizes, int n_in,
                              void* d_out, int out_size, void* d_ws, size_t ws_size,
                              hipStream_t stream) {
  const float* x   = (const float*)d_in[0];
  const float* adj = (const float*)d_in[1];
  const float* w1  = (const float*)d_in[2];
  const float* b1  = (const float*)d_in[3];
  const float* w2  = (const float*)d_in[4];
  const float* b2  = (const float*)d_in[5];
  const float* wg  = (const float*)d_in[6];
  const float* bg  = (const float*)d_in[7];
  const float* wd  = (const float*)d_in[8];
  const float* bd  = (const float*)d_in[9];
  const float* wp1 = (const float*)d_in[10];
  const float* bp1 = (const float*)d_in[11];
  const float* wp2 = (const float*)d_in[12];
  const float* bp2 = (const float*)d_in[13];
  const float* wo  = (const float*)d_in[14];
  const float* bo  = (const float*)d_in[15];
  float* out = (float*)d_out;

  char* ws = (char*)d_ws;
  size_t off = 0;
  auto alloc = [&](size_t bytes) {
    char* p = ws + off;
    off += (bytes + 255) & ~(size_t)255;
    return p;
  };
  float*    h1      = (float*)alloc((size_t)NN * 64 * 4);
  float*    h       = (float*)alloc((size_t)NN * 128 * 4);
  float*    partial = (float*)alloc((size_t)32 * NN * 4);
  float*    dinv    = (float*)alloc((size_t)NN * 4);
  unsigned* bits    = (unsigned*)alloc((size_t)NN * NW * 4);            // 32 MB
  float*    zf      = (float*)alloc((size_t)NN * 128 * 4);
  __bf16*   zhiT    = (__bf16*)alloc((size_t)NN * 128 * 2);
  __bf16*   zloT    = (__bf16*)alloc((size_t)NN * 128 * 2);
  float*    spart   = (float*)alloc((size_t)JSPLIT * NN * 128 * 4);     // 64 MB
  float*    xg      = (float*)alloc((size_t)NN * 128 * 4);
  float*    xd      = (float*)alloc((size_t)NN * 128 * 4);
  float*    p1      = (float*)alloc((size_t)NN * 128 * 4);
  float*    p2      = (float*)alloc((size_t)NN * 64 * 4);
  (void)ws_size; (void)in_sizes; (void)n_in; (void)out_size;

  // encoder MLP
  lin_kernel<32, 64, 4, 8, true><<<NN / 32, 256, 0, stream>>>(x, w1, b1, h1);
  lin_kernel<64, 128, 2, 8, true><<<NN / 16, 256, 0, stream>>>(h1, w2, b2, h);
  // deg/dinv + adjacency bit-pack (the single full 1 GiB adjacency read)
  colsum_pack_kernel<<<dim3(NN / 256, 32), 256, 0, stream>>>(adj, partial, bits);
  dinv_kernel<<<NN / 256, 256, 0, stream>>>(partial, dinv);
  // z = dinv * (h @ wg), f32 + transposed bf16 hi/lo
  zgen_kernel<<<NN / 16, 256, 0, stream>>>(h, wg, dinv, zf, zhiT, zloT);
  // A_hat^T z via bf16 WMMA over the bitmask, j-split x8, then reduce+epilogue
  spmm_wmma_kernel<<<dim3(NN / 32, JSPLIT), 64, 0, stream>>>(bits, zhiT, zloT, spart);
  spmm_reduce_kernel<<<(NN * 128) / 256, 256, 0, stream>>>(spart, zf, dinv, bg, xg);
  // decoder chain
  lin_kernel<128, 128, 2, 8, true><<<NN / 16, 256, 0, stream>>>(xg, wd, bd, xd);
  concat_lin_kernel<<<NN / 16, 256, 0, stream>>>(xd, h, wp1, bp1, p1);
  lin_kernel<128, 64, 4, 8, true><<<NN / 32, 256, 0, stream>>>(p1, wp2, bp2, p2);
  out_kernel<<<NN / 256, 256, 0, stream>>>(p2, wo, bo, out);
}
